// AE_KGCN_17712445129477
// MI455X (gfx1250) — matrix operand order repaired and checked by reading, not compiled, not run on hardware
//
#include <hip/hip_runtime.h>
#include <stdint.h>

// ---------------- problem constants ----------------
#define Bsz   64
#define NV    25274
#define NVP   25280        // NV padded to multiple of 32 (WMMA K depth)
#define NITEM 25000
#define DIM   32
#define NN    4
#define H1c   512
#define H2c   64
#define KSPLIT 10
#define KCH    2528        // NVP / KSPLIT, = 79 * 32 (no K tail)

typedef __bf16 bf16;
typedef __attribute__((ext_vector_type(16))) __bf16 v16bf;
typedef __attribute__((ext_vector_type(8)))  __bf16 v8bf;
typedef __attribute__((ext_vector_type(8)))  float  v8f;
typedef __attribute__((ext_vector_type(4)))  unsigned v4u;
typedef __attribute__((ext_vector_type(8)))  int      v8i;
typedef __attribute__((ext_vector_type(4)))  int      v4i;

#define HAVE_TDM (__has_builtin(__builtin_amdgcn_tensor_load_to_lds) && \
                  __has_builtin(__builtin_amdgcn_s_wait_tensorcnt))

// ---------------- helpers ----------------
__device__ __forceinline__ bf16 f2bf(float f) {
  unsigned u = __builtin_bit_cast(unsigned, f);
  unsigned r = u + 0x7FFFu + ((u >> 16) & 1u);   // round-to-nearest-even
  unsigned short h = (unsigned short)(r >> 16);
  return __builtin_bit_cast(bf16, h);
}

__device__ __forceinline__ float selu_f(float x) {
  const float sc = 1.0507009873554805f, al = 1.6732632423543772f;
  return x > 0.f ? sc * x : sc * al * (__expf(x) - 1.f);
}

// A fragment: 16x32 bf16, row-major source (contiguous k = 16*(lane/16) + s, same order as B).
__device__ __forceinline__ v16bf ld_afrag(const bf16* base, int ld, int row0, int k0, int lane) {
  const bf16* p = base + (size_t)(row0 + (lane & 15)) * ld + k0 + ((lane >> 4) << 4);
  v8bf lo = *(const v8bf*)p;          // K +0..7  (+16 for hi lanes)
  v8bf hi = *(const v8bf*)(p + 8);    // K +8..15 (+16 for hi lanes)
  v16bf a;
#pragma unroll
  for (int i = 0; i < 8; ++i) { a[i] = lo[i]; a[i + 8] = hi[i]; }
  return a;
}

// B fragment: 32x16 bf16 from B-transposed source (srcT[n][k], K contiguous per lane).
__device__ __forceinline__ v16bf ld_bfragT(const bf16* baseT, int ld, int n0, int k0, int lane) {
  const bf16* p = baseT + (size_t)(n0 + (lane & 15)) * ld + k0 + ((lane >> 4) << 4);
  v8bf lo = *(const v8bf*)p;
  v8bf hi = *(const v8bf*)(p + 8);
  v16bf b;
#pragma unroll
  for (int i = 0; i < 8; ++i) { b[i] = lo[i]; b[i + 8] = hi[i]; }
  return b;
}

// B fragment from a K-major LDS panel tile[k][16] (strided u16 gathers).
__device__ __forceinline__ v16bf ld_bfrag_panel(const bf16* tile, int k0, int lane) {
  int n = lane & 15;
  int kb = k0 + ((lane >> 4) << 4);
  v16bf b;
#pragma unroll
  for (int i = 0; i < 16; ++i) b[i] = tile[(kb + i) * 16 + n];
  return b;
}

#define WMMA_BF16(a, b, c) \
  __builtin_amdgcn_wmma_f32_16x16x32_bf16(false, (a), false, (b), (short)0, (c), false, false)

// ---------------- kernels ----------------

// fp32 -> bf16 with column padding (zero fill)
__global__ void k_cvt(const float* __restrict__ src, bf16* __restrict__ dst,
                      int rows, int cols, int dstride) {
  size_t total = (size_t)rows * dstride;
  for (size_t idx = (size_t)blockIdx.x * 256 + threadIdx.x; idx < total;
       idx += (size_t)gridDim.x * 256) {
    int r = (int)(idx / dstride), c = (int)(idx % dstride);
    float v = (c < cols) ? src[(size_t)r * cols + c] : 0.f;
    dst[idx] = f2bf(v);
  }
}

// encoder GEMM, split-K: parts[z][64][512] += x_bf @ enc_w0_bf^T over K slice z.
// grid (4 m-tiles, 32 n-tiles, KSPLIT), 1 wave per block.
__global__ void __launch_bounds__(32)
k_gemm_enc(const bf16* __restrict__ xbf, const bf16* __restrict__ w0bf,
           float* __restrict__ parts) {
  int lane = threadIdx.x;
  int m0 = blockIdx.x * 16, n0 = blockIdx.y * 16;
  int kbeg = blockIdx.z * KCH, kend = kbeg + KCH;
  const bf16* pfa = xbf  + (size_t)(m0 + (lane & 15)) * NVP;
  const bf16* pfb = w0bf + (size_t)(n0 + (lane & 15)) * NVP;
  v8f c = {};
  for (int k0 = kbeg; k0 < kend; k0 += 32) {
    __builtin_prefetch(pfa + k0 + 128, 0, 1);   // global_prefetch_b8 (stream ahead)
    __builtin_prefetch(pfb + k0 + 128, 0, 1);
    v16bf a = ld_afrag(xbf, NVP, m0, k0, lane);
    v16bf b = ld_bfragT(w0bf, NVP, n0, k0, lane);
    c = WMMA_BF16(a, b, c);
  }
  int n  = n0 + (lane & 15);
  int mb = m0 + ((lane >> 4) << 3);
  float* po = parts + (size_t)blockIdx.z * Bsz * H1c;
#pragma unroll
  for (int i = 0; i < 8; ++i) po[(mb + i) * H1c + n] = c[i];
}

// reduce K-split partials, add bias, selu -> h1
__global__ void __launch_bounds__(256)
k_h1_reduce(const float* __restrict__ parts, const float* __restrict__ b0,
            float* __restrict__ h1) {
  int idx = blockIdx.x * 256 + threadIdx.x;
  if (idx < Bsz * H1c) {
    float s = b0[idx % H1c];
#pragma unroll
    for (int p = 0; p < KSPLIT; ++p) s += parts[(size_t)p * Bsz * H1c + idx];
    h1[idx] = selu_f(s);
  }
}

// h2 = selu(h1 @ enc_w1^T + enc_b1); user = h2 @ u_w^T + u_b   (tiny, 1 block)
__global__ void __launch_bounds__(256)
k_h2_user(const float* __restrict__ h1, const float* __restrict__ w1,
          const float* __restrict__ b1, const float* __restrict__ uw,
          const float* __restrict__ ub, float* __restrict__ h2o,
          float* __restrict__ usero) {
  __shared__ float h2s[Bsz][H2c];
  int t = threadIdx.x;
  for (int idx = t; idx < Bsz * H2c; idx += 256) {
    int b = idx / H2c, j = idx % H2c;
    float s = b1[j];
    for (int k = 0; k < H1c; ++k) s += h1[b * H1c + k] * w1[j * H1c + k];
    float v = selu_f(s);
    h2s[b][j] = v;
    h2o[idx]  = v;
  }
  __syncthreads();
  for (int idx = t; idx < Bsz * DIM; idx += 256) {
    int b = idx / DIM, d = idx % DIM;
    float s = ub[d];
    for (int k = 0; k < H2c; ++k) s += h2s[b][k] * uw[d * H2c + k];
    usero[idx] = s;
  }
}

// z = selu(h2 @ enc_w1 + dec_b0); batch-norm over batch; emit bf16 z (1 block, 512 thr)
__global__ void __launch_bounds__(512)
k_z_bn(const float* __restrict__ h2, const float* __restrict__ w1,
       const float* __restrict__ db0, const float* __restrict__ gam,
       const float* __restrict__ bet, bf16* __restrict__ zbf) {
  __shared__ float h2s[Bsz][H2c];
  int t = threadIdx.x;
  for (int idx = t; idx < Bsz * H2c; idx += 512) h2s[idx / H2c][idx % H2c] = h2[idx];
  __syncthreads();
  int f = t;
  float bias = db0[f];
  float s = 0.f, s2 = 0.f;
  for (int b = 0; b < Bsz; ++b) {
    float z = bias;
    for (int k = 0; k < H2c; ++k) z += h2s[b][k] * w1[k * H1c + f];
    z = selu_f(z);
    s += z; s2 += z * z;
  }
  float mu  = s * (1.f / Bsz);
  float var = s2 * (1.f / Bsz) - mu * mu;        // biased (torch BN training)
  float inv = rsqrtf(var + 1e-5f) * gam[f];
  float be  = bet[f];
  for (int b = 0; b < Bsz; ++b) {
    float z = bias;
    for (int k = 0; k < H2c; ++k) z += h2s[b][k] * w1[k * H1c + f];
    z = selu_f(z);
    zbf[b * H1c + f] = f2bf((z - mu) * inv + be);
  }
}

// KGCN: one block per item. scores/softmax/agg scalar, fc layer via WMMA,
// ret[b,m] = dot(tanh(cat@fc_w^T+fc_b), user).
__global__ void __launch_bounds__(256)
k_kgcn(const float* __restrict__ user, const bf16* __restrict__ fcwbf,
       const float* __restrict__ fcb, const float* __restrict__ item_emb,
       const float* __restrict__ nbr_e, const float* __restrict__ nbr_r,
       float* __restrict__ ret) {
  __shared__ float us[Bsz][DIM];
  __shared__ float sc[Bsz][NN];
  __shared__ bf16  cat[Bsz][2 * DIM];
  __shared__ float ent[Bsz][DIM];
  int t = threadIdx.x, lane = t & 31, wv = t >> 5;
  int m = blockIdx.x;

  for (int idx = t; idx < Bsz * DIM; idx += 256) us[idx / DIM][idx % DIM] = user[idx];
  __syncthreads();

  { // user-relation scores (mean over DIM)
    int b = t & 63, g = t >> 6;
    const float* rp = nbr_r + (size_t)(m * NN + g) * DIM;
    float s = 0.f;
    for (int d = 0; d < DIM; ++d) s += us[b][d] * rp[d];
    sc[b][g] = s * (1.f / DIM);
  }
  __syncthreads();
  if (t < Bsz) { // softmax over 4 neighbors
    float a0 = sc[t][0], a1 = sc[t][1], a2 = sc[t][2], a3 = sc[t][3];
    float mx = fmaxf(fmaxf(a0, a1), fmaxf(a2, a3));
    float e0 = __expf(a0 - mx), e1 = __expf(a1 - mx), e2 = __expf(a2 - mx), e3 = __expf(a3 - mx);
    float is = 1.f / (e0 + e1 + e2 + e3);
    sc[t][0] = e0 * is; sc[t][1] = e1 * is; sc[t][2] = e2 * is; sc[t][3] = e3 * is;
  }
  __syncthreads();
  { // weighted neighbor aggregation + build cat = [item_emb | agg] (bf16)
    int b = t & 63, g = t >> 6;
    float w0 = sc[b][0], w1 = sc[b][1], w2 = sc[b][2], w3 = sc[b][3];
    const float* ep = nbr_e + (size_t)m * NN * DIM;
#pragma unroll
    for (int q = 0; q < 8; ++q) {
      int d = g * 8 + q;
      float a = w0 * ep[d] + w1 * ep[DIM + d] + w2 * ep[2 * DIM + d] + w3 * ep[3 * DIM + d];
      cat[b][DIM + d] = f2bf(a * 0.25f);
      cat[b][d]       = f2bf(item_emb[(size_t)m * DIM + d]);
    }
  }
  __syncthreads();
  { // ent = tanh(cat @ fc_w^T + fc_b): M=64,N=32,K=64 -> 8 wave-tiles, 2 WMMA each
    int m0 = (wv & 3) * 16, j0 = (wv >> 2) * 16;
    v16bf b0 = ld_bfragT(fcwbf, 2 * DIM, j0, 0, lane);
    v16bf b1 = ld_bfragT(fcwbf, 2 * DIM, j0, 32, lane);
    v16bf a0 = ld_afrag(&cat[0][0], 2 * DIM, m0, 0, lane);
    v16bf a1 = ld_afrag(&cat[0][0], 2 * DIM, m0, 32, lane);
    v8f c = {};
    c = WMMA_BF16(a0, b0, c);
    c = WMMA_BF16(a1, b1, c);
    int j  = j0 + (lane & 15);
    int mb = m0 + ((lane >> 4) << 3);
    float bias = fcb[j];
#pragma unroll
    for (int i = 0; i < 8; ++i) ent[mb + i][j] = tanhf(c[i] + bias);
  }
  __syncthreads();
  if (t < Bsz) { // ret[b,m] = dot(ent[b], user[b])
    float s = 0.f;
    for (int d = 0; d < DIM; ++d) s += ent[t][d] * us[t][d];
    ret[(size_t)t * NITEM + m] = s;
  }
}

// out = sigmoid(z_bn @ enc_w0 + dec_b1 + ret)  [64, 25274]
// grid (4 m-tiles, 1580 n-tiles), 1 wave per block.
// B panel [512 k x 16 n] of enc_w0_bf staged to LDS via the Tensor Data Mover
// (one TENSOR_LOAD_TO_LDS per block, D# per ISA ch.8), tracked by TENSORcnt.
__global__ void __launch_bounds__(32)
k_dec(const bf16* __restrict__ zbf, const bf16* __restrict__ w0bf,
      const float* __restrict__ db1, const float* __restrict__ ret,
      float* __restrict__ out) {
  __shared__ bf16 bpanel[H1c][16];     // [k][n] tile, 16 KB
  int lane = threadIdx.x;
  int m0 = blockIdx.x * 16, n0 = blockIdx.y * 16;

#if HAVE_TDM
  {
    // D# group 0: count=1 (user), lds_addr, 57-bit global tile address, type=2
    unsigned lds_addr = (unsigned)(uintptr_t)&bpanel[0][0];   // low 32b of generic = LDS offset
    unsigned long long ga = (unsigned long long)(uintptr_t)(w0bf + n0);
    v4u g0 = { 1u,
               lds_addr,
               (unsigned)(ga & 0xFFFFFFFFu),
               (unsigned)((ga >> 32) & 0x01FFFFFFu) | (2u << 30) };
    // D# group 1: data_size=2B, tensor 25280 x 512, tile 16 x 512, dim0 stride = 25280
    unsigned td0 = NVP, td1 = H1c, t0 = 16, t1 = H1c, t2 = 0;
    unsigned long long s0 = NVP;
    v8i g1 = { (int)(1u << 16),
               (int)((td0 & 0xFFFFu) << 16),
               (int)(((td0 >> 16) & 0xFFFFu) | ((td1 & 0xFFFFu) << 16)),
               (int)(((td1 >> 16) & 0xFFFFu) | (t0 << 16)),
               (int)(t1 | (t2 << 16)),
               (int)(s0 & 0xFFFFFFFFu),
               (int)((s0 >> 32) & 0xFFFFu),
               0 };
    v4i g2 = { 0, 0, 0, 0 };
    v4i g3 = { 0, 0, 0, 0 };
    v8i g4 = { 0, 0, 0, 0, 0, 0, 0, 0 };   // unused trailing group (2D tensor)
    __builtin_amdgcn_tensor_load_to_lds(g0, g1, g2, g3, g4, 0);
    __builtin_amdgcn_s_wait_tensorcnt(0);    // TDM complete -> LDS panel valid
  }
#else
  // fallback: cooperative staging, 32 lanes x 16 rows each (32B per row)
  for (int r = lane; r < H1c; r += 32) {
    const bf16* p = w0bf + (size_t)r * NVP + n0;
    v8bf r0 = *(const v8bf*)p;
    v8bf r1 = *(const v8bf*)(p + 8);
    *(v8bf*)&bpanel[r][0] = r0;
    *(v8bf*)&bpanel[r][8] = r1;
  }
  __syncthreads();
#endif

  v8f c = {};
  for (int k0 = 0; k0 < H1c; k0 += 32) {
    v16bf a = ld_afrag(zbf, H1c, m0, k0, lane);
    v16bf b = ld_bfrag_panel(&bpanel[0][0], k0, lane);
    c = WMMA_BF16(a, b, c);
  }
  int n  = n0 + (lane & 15);
  int mb = m0 + ((lane >> 4) << 3);
  if (n < NV) {
    float bias = db1[n];
#pragma unroll
    for (int i = 0; i < 8; ++i) {
      float r = (n < NITEM) ? ret[(size_t)(mb + i) * NITEM + n] : 0.f;
      float v = c[i] + bias + r;
      out[(size_t)(mb + i) * NV + n] = 1.f / (1.f + __expf(-v));
    }
  }
}

// ---------------- launch ----------------
extern "C" void kernel_launch(void* const* d_in, const int* in_sizes, int n_in,
                              void* d_out, int out_size, void* d_ws, size_t ws_size,
                              hipStream_t stream) {
  const float* x        = (const float*)d_in[0];
  const float* enc_w0   = (const float*)d_in[1];
  const float* enc_b0   = (const float*)d_in[2];
  const float* enc_w1   = (const float*)d_in[3];
  const float* enc_b1   = (const float*)d_in[4];
  const float* dec_b0   = (const float*)d_in[5];
  const float* dec_b1   = (const float*)d_in[6];
  const float* bn_gamma = (const float*)d_in[7];
  const float* bn_beta  = (const float*)d_in[8];
  const float* u_w      = (const float*)d_in[9];
  const float* u_b      = (const float*)d_in[10];
  const float* fc_w     = (const float*)d_in[11];
  const float* fc_b     = (const float*)d_in[12];
  const float* item_emb = (const float*)d_in[13];
  const float* nbr_ent  = (const float*)d_in[14];
  const float* nbr_rel  = (const float*)d_in[15];
  float* out = (float*)d_out;

  // workspace carve-up (all chunks 256B-aligned sizes)
  char* w = (char*)d_ws;
  bf16*  xbf   = (bf16*)w;   w += (size_t)Bsz * NVP * 2;          // 3,235,840
  bf16*  w0bf  = (bf16*)w;   w += (size_t)H1c * NVP * 2;          // 25,886,720
  bf16*  fcwbf = (bf16*)w;   w += (size_t)DIM * 2 * DIM * 2;      // 4,096
  float* h1p   = (float*)w;  w += (size_t)KSPLIT * Bsz * H1c * 4; // 1,310,720
  float* h1    = (float*)w;  w += (size_t)Bsz * H1c * 4;          // 131,072
  float* h2    = (float*)w;  w += (size_t)Bsz * H2c * 4;          // 16,384
  float* user  = (float*)w;  w += (size_t)Bsz * DIM * 4;          // 8,192
  bf16*  zbf   = (bf16*)w;   w += (size_t)Bsz * H1c * 2;          // 65,536
  float* retw  = (float*)w;  w += (size_t)Bsz * NITEM * 4;        // 6,400,000

  // 1) bf16 conversions (zero-padded K)
  k_cvt<<<4096, 256, 0, stream>>>(x,      xbf,   Bsz, NV, NVP);
  k_cvt<<<4096, 256, 0, stream>>>(enc_w0, w0bf,  H1c, NV, NVP);
  k_cvt<<<16,   256, 0, stream>>>(fc_w,   fcwbf, DIM, 2 * DIM, 2 * DIM);

  // 2) encoder big GEMM: split-K WMMA partials, then reduce + bias + selu
  k_gemm_enc<<<dim3(4, 32, KSPLIT), 32, 0, stream>>>(xbf, w0bf, h1p);
  k_h1_reduce<<<(Bsz * H1c + 255) / 256, 256, 0, stream>>>(h1p, enc_b0, h1);

  // 3) small chain: h2, user latent
  k_h2_user<<<1, 256, 0, stream>>>(h1, enc_w1, enc_b1, u_w, u_b, h2, user);

  // 4) tied decode first layer + batch-norm -> z (bf16)
  k_z_bn<<<1, 512, 0, stream>>>(h2, enc_w1, dec_b0, bn_gamma, bn_beta, zbf);

  // 5) KGCN aggregator + fc (WMMA) + per-item scores
  k_kgcn<<<NITEM, 256, 0, stream>>>(user, fcwbf, fc_b, item_emb, nbr_ent, nbr_rel, retw);

  // 6) decoder big GEMM (TDM-staged B panel + WMMA), fused +dec_b1 +ret, sigmoid
  k_dec<<<dim3(4, 1580), 32, 0, stream>>>(zbf, w0bf, dec_b1, retw, out);
}
